// CausalGraphGenerator_59803124629641
// MI455X (gfx1250) — matrix operand order repaired and checked by reading, not compiled, not run on hardware
//
#include <hip/hip_runtime.h>
#include <hip/hip_bf16.h>

typedef __attribute__((ext_vector_type(16))) _Float16 v16h;
typedef __attribute__((ext_vector_type(8)))  float    v8f;

#define BATCH 4
#define WLEN  2048
#define CHAN  64
#define HID   128
#define NSER  (BATCH * CHAN)   // 256 folded series

// exact-gelu derivative: Phi(z) + z*phi(z)
__device__ __forceinline__ float dgelu(float z) {
    const float kInvSqrt2   = 0.70710678118654752f;
    const float kInvSqrt2Pi = 0.39894228040143268f;
    float cdf = 0.5f * (1.0f + erff(z * kInvSqrt2));
    float pdf = kInvSqrt2Pi * __expf(-0.5f * z * z);
    return cdf + z * pdf;
}

__global__ void cgg_zero_ws(float* ws) {
    int i = blockIdx.x * blockDim.x + threadIdx.x;
    if (i < NSER) ws[i] = 0.0f;
}

// One block = 4 waves; block handles one series n and 64 consecutive w's
// (one 16-w tile per wave). Grid.x = NSER * (WLEN/64) = 8192.
__global__ void __launch_bounds__(128)
cgg_grad_kernel(const float* __restrict__ X, const float* __restrict__ w1,
                const float* __restrict__ b1, const float* __restrict__ w2,
                float* __restrict__ ws) {
    __shared__ float    xs[68];                 // x[gw0-2 .. gw0+65], zero-padded
    __shared__ _Float16 w2h[HID];
    __shared__ _Float16 u_lds[4][16][HID + 8];  // per-wave u tile, padded rows

    const int bx   = blockIdx.x;
    const int n    = bx >> 5;        // series id (b*CHAN + c)
    const int grp  = bx & 31;        // 64-wide w group
    const int wave = threadIdx.x >> 5;
    const int lane = threadIdx.x & 31;
    const int b    = n >> 6;
    const int c    = n & (CHAN - 1);
    const int gw0  = grp * 64;
    const int w0   = gw0 + wave * 16;

    const float* xp = X + (size_t)b * WLEN * CHAN + c;  // x[w] = xp[w*CHAN]

    // Stage the x window and f16 weights for conv2
    {
        int t = threadIdx.x;
        if (t < 68) {
            int gw = gw0 - 2 + t;
            xs[t] = (gw >= 0 && gw < WLEN) ? xp[(size_t)gw * CHAN] : 0.0f;
        }
        if (t < HID) w2h[t] = (_Float16)w2[t];
    }
    __syncthreads();

    // Phase A: u[k,w] = w1[k,2]*gp(w-1) + w1[k,1]*gp(w) + w1[k,0]*gp(w+1)
    // Each lane owns 4 k-values; rolling window over 18 gelu' evals per k.
    const int lbase = wave * 16 + 2;  // xs index of x[w0]
    #pragma unroll
    for (int j = 0; j < 4; ++j) {
        const int k = lane + 32 * j;
        const float a0 = w1[3 * k + 0];
        const float a1 = w1[3 * k + 1];
        const float a2 = w1[3 * k + 2];
        const float bi = b1[k];

        auto gpw = [&](int t) -> float {   // t = w' - w0
            int wp = w0 + t;
            if ((unsigned)wp >= (unsigned)WLEN) return 0.0f;  // invalid conv1 row
            float z = a0 * xs[lbase + t - 1] + a1 * xs[lbase + t]
                    + a2 * xs[lbase + t + 1] + bi;
            return dgelu(z);
        };

        float gm = gpw(-1);
        float g0 = gpw(0);
        #pragma unroll 4
        for (int t = 0; t < 16; ++t) {
            float gp1 = gpw(t + 1);
            float uu  = a2 * gm + a1 * g0 + a0 * gp1;
            u_lds[wave][t][k] = (_Float16)uu;
            gm = g0; g0 = gp1;
        }
    }
    __syncthreads();

    // Phase B: g[w0+m] = sum_k u[k, w0+m] * w2[k] via 4 chained
    // v_wmma_f32_16x16x32_f16. B holds w2 replicated in all 16 columns,
    // so every D column equals the matvec result.
    // A 16x32 f16 layout: lane<16 -> row m=lane, K in {0..7, 16..23};
    //                     lane>=16 -> row m=lane-16, K in {8..15, 24..31}.
    // B 32x16 f16 layout: lanes 0-15 hold K {0..7, 16..23}, lanes 16-31 hold K {8..15, 24..31}.
    const int m     = lane & 15;
    const int aksel = (lane & 16) ? 8  : 0;
    const int bksel = (lane & 16) ? 8  : 0;

    v8f acc = {};
    #pragma unroll
    for (int kb = 0; kb < HID; kb += 32) {
        v16h af, bf;
        #pragma unroll
        for (int e = 0; e < 8; ++e) {
            af[e]     = u_lds[wave][m][kb + aksel + e];
            af[e + 8] = u_lds[wave][m][kb + aksel + 16 + e];
        }
        #pragma unroll
        for (int e = 0; e < 8; ++e) {
            bf[e]     = w2h[kb + bksel + e];
            bf[e + 8] = w2h[kb + bksel + 16 + e];
        }
        acc = __builtin_amdgcn_wmma_f32_16x16x32_f16(
            /*neg_a=*/false, af, /*neg_b=*/false, bf,
            /*c_mod=*/(short)0, acc, /*reuse_a=*/false, /*reuse_b=*/false);
    }

    // D layout: lane 0 (N=0) holds rows M=0..7, lane 16 holds M=8..15.
    float s = 0.0f;
    #pragma unroll
    for (int r = 0; r < 8; ++r) s += fabsf(acc[r]);
    if ((lane & 15) == 0) atomicAdd(&ws[n], s);
}

// Literal tail of the reference: diag_vals -> A -> A - A^T -> relu(.-h) -> mask.
__global__ void cgg_finalize(const float* __restrict__ ws,
                             const float* __restrict__ hptr,
                             float* __restrict__ out) {
    int idx = blockIdx.x * blockDim.x + threadIdx.x;
    if (idx >= BATCH * CHAN * CHAN) return;
    int b = idx / (CHAN * CHAN);
    int r = idx - b * (CHAN * CHAN);
    int i = r / CHAN;            // src
    int j = r - i * CHAN;        // tgt
    const float inv_w = 1.0f / (float)WLEN;
    float dvi = ws[b * CHAN + i] * inv_w;
    float dvj = ws[b * CHAN + j] * inv_w;
    // A[b,i,j] = dv[i]*delta_ij ; A_diff = A - A^T
    float adiff = (i == j) ? (dvi - dvj) : 0.0f;
    float v = adiff - hptr[0];
    v = v > 0.0f ? v : 0.0f;
    if (i == j) v = 0.0f;
    out[idx] = v;
}

extern "C" void kernel_launch(void* const* d_in, const int* in_sizes, int n_in,
                              void* d_out, int out_size, void* d_ws, size_t ws_size,
                              hipStream_t stream) {
    (void)in_sizes; (void)n_in; (void)out_size; (void)ws_size;
    const float* X  = (const float*)d_in[0];
    const float* w1 = (const float*)d_in[1];
    const float* b1 = (const float*)d_in[2];
    const float* w2 = (const float*)d_in[3];
    // d_in[4] = b2 (not needed for the gradient), d_in[5] = h
    const float* h  = (const float*)d_in[5];
    float* out = (float*)d_out;
    float* ws  = (float*)d_ws;   // NSER floats of |g| sums

    cgg_zero_ws<<<1, NSER, 0, stream>>>(ws);
    cgg_grad_kernel<<<NSER * (WLEN / 64), 128, 0, stream>>>(X, w1, b1, w2, ws);
    cgg_finalize<<<(BATCH * CHAN * CHAN + 255) / 256, 256, 0, stream>>>(ws, h, out);
}
